// GAT_1812476199283
// MI455X (gfx1250) — compile-verified, matrix operand order checked by reading, not run on hardware
//
#include <hip/hip_runtime.h>
#include <math.h>

#define IN_C   165
#define KPAD   192            // 6 chunks of 32
#define HID    256
#define HEADS  8
#define C1     32             // HID / HEADS
#define OUT_C  2
#define NEG_SLOPE 0.2f

typedef __attribute__((ext_vector_type(16))) _Float16 v16h;
typedef __attribute__((ext_vector_type(8)))  _Float16 half8;
typedef __attribute__((ext_vector_type(8)))  float    v8f;

// ---------------------------------------------------------------- utilities

__device__ __forceinline__ void atomicMaxF(float* addr, float v) {
    // signed-max for v>=0, unsigned-min for v<0; init value must be -inf
    if (v >= 0.0f) atomicMax((int*)addr, __float_as_int(v));
    else           atomicMin((unsigned int*)addr, __float_as_uint(v));
}

__device__ __forceinline__ void edge_sd(const long long* __restrict__ ei,
                                        long long E, long long e,
                                        int& s, int& d) {
    if (e < E) { s = (int)ei[e]; d = (int)ei[E + e]; }
    else       { s = d = (int)(e - E); }           // appended self-loops
}

__global__ void gat_fill(float* __restrict__ p, float v, long long n) {
    long long i = (long long)blockIdx.x * blockDim.x + threadIdx.x;
    if (i < n) p[i] = v;
}

// ------------------------------------------------- f32 -> f16 staging (GEMM1)

__global__ void gat_cvt_x(const float* __restrict__ x, _Float16* __restrict__ xh,
                          int nrows, long long total) {
    long long i = (long long)blockIdx.x * blockDim.x + threadIdx.x;
    if (i >= total) return;
    int row = (int)(i / KPAD), k = (int)(i % KPAD);
    float v = (row < nrows && k < IN_C) ? x[(size_t)row * IN_C + k] : 0.0f;
    xh[i] = (_Float16)v;
}

__global__ void gat_cvt_w(const float* __restrict__ W1, _Float16* __restrict__ wt) {
    long long i = (long long)blockIdx.x * blockDim.x + threadIdx.x;   // over HID*KPAD
    if (i >= (long long)HID * KPAD) return;
    int n = (int)(i / KPAD), k = (int)(i % KPAD);
    float v = (k < IN_C) ? W1[(size_t)k * HID + n] : 0.0f;            // transpose
    wt[i] = (_Float16)v;
}

// ----------------------------------------------------- GEMM1: h1 = x @ W1
// one wave per 16x16 tile of [N x 256]; K loop = 6 chunks of 32 (f16 WMMA)

__global__ void __launch_bounds__(32)
gat_gemm1(const _Float16* __restrict__ xh, const _Float16* __restrict__ wt,
          float* __restrict__ h1, int nrows) {
    const int mtile = blockIdx.x;
    const int ntile = blockIdx.y;
    const int lane  = threadIdx.x;
    const int l15   = lane & 15;
    const int half  = lane >> 4;

    const _Float16* ap = xh + (size_t)(mtile * 16 + l15) * KPAD + 8 * half;
    const _Float16* bp = wt + (size_t)(ntile * 16 + l15) * KPAD + 8 * half;

    v8f c = {};
#pragma unroll
    for (int kc = 0; kc < KPAD / 32; ++kc) {
        half8 alo = *(const half8*)(ap + kc * 32);
        half8 ahi = *(const half8*)(ap + kc * 32 + 16);
        half8 blo = *(const half8*)(bp + kc * 32);
        half8 bhi = *(const half8*)(bp + kc * 32 + 16);
        v16h a = __builtin_shufflevector(alo, ahi, 0,1,2,3,4,5,6,7,8,9,10,11,12,13,14,15);
        v16h b = __builtin_shufflevector(blo, bhi, 0,1,2,3,4,5,6,7,8,9,10,11,12,13,14,15);
        c = __builtin_amdgcn_wmma_f32_16x16x32_f16(false, a, false, b, (short)0, c,
                                                   false, false);
    }

    float* dp = h1 + (size_t)(mtile * 16 + 8 * half) * HID + ntile * 16 + l15;
    if (mtile * 16 + 16 <= nrows) {
        // wave-uniform fast path: whole tile in range, no exec-mask churn
#pragma unroll
        for (int r = 0; r < 8; ++r) dp[(size_t)r * HID] = c[r];
    } else {
#pragma unroll
        for (int r = 0; r < 8; ++r) {
            if (mtile * 16 + r + 8 * half < nrows) dp[(size_t)r * HID] = c[r];
        }
    }
}

// ----------------------------------------- attention logits per node (layer 1)

__global__ void gat_att1(const float* __restrict__ h1,
                         const float* __restrict__ att_src, const float* __restrict__ att_dst,
                         float* __restrict__ asrc, float* __restrict__ adst, int n) {
    long long gid = (long long)blockIdx.x * blockDim.x + threadIdx.x;
    if (gid >= (long long)n * HEADS) return;
    int node = (int)(gid >> 3), h = (int)(gid & 7);
    const float* hp = h1 + (size_t)node * HID + h * C1;
    const float* ws = att_src + h * C1;
    const float* wd = att_dst + h * C1;
    float s = 0.0f, d = 0.0f;
#pragma unroll
    for (int c = 0; c < C1; ++c) { s += hp[c] * ws[c]; d += hp[c] * wd[c]; }
    asrc[gid] = s; adst[gid] = d;
}

// --------------------------------------------- layer-1 segment softmax passes

__global__ void gat_e1_max(const long long* __restrict__ ei, long long E, long long Et,
                           const float* __restrict__ asrc, const float* __restrict__ adst,
                           float* __restrict__ emax) {
    long long gid = (long long)blockIdx.x * blockDim.x + threadIdx.x;
    if (gid >= Et * HEADS) return;
    long long e = gid >> 3; int h = (int)(gid & 7);
    int s, d; edge_sd(ei, E, e, s, d);
    float v = asrc[(size_t)s * HEADS + h] + adst[(size_t)d * HEADS + h];
    v = v > 0.0f ? v : NEG_SLOPE * v;
    atomicMaxF(&emax[(size_t)d * HEADS + h], v);
}

__global__ void gat_e1_sum(const long long* __restrict__ ei, long long E, long long Et,
                           const float* __restrict__ asrc, const float* __restrict__ adst,
                           const float* __restrict__ emax, float* __restrict__ esum) {
    long long gid = (long long)blockIdx.x * blockDim.x + threadIdx.x;
    if (gid >= Et * HEADS) return;
    long long e = gid >> 3; int h = (int)(gid & 7);
    int s, d; edge_sd(ei, E, e, s, d);
    float v = asrc[(size_t)s * HEADS + h] + adst[(size_t)d * HEADS + h];
    v = v > 0.0f ? v : NEG_SLOPE * v;
    atomicAdd(&esum[(size_t)d * HEADS + h], __expf(v - emax[(size_t)d * HEADS + h]));
}

// one 256-thread block per edge: coalesced gather of h1[src], scatter-add to out[dst]
__global__ void __launch_bounds__(256)
gat_e1_agg(const long long* __restrict__ ei, long long E, long long Et,
           const float* __restrict__ asrc, const float* __restrict__ adst,
           const float* __restrict__ emax, const float* __restrict__ esum,
           const float* __restrict__ h1, float* __restrict__ out1) {
    long long e = blockIdx.x;
    if (e >= Et) return;
    int t = threadIdx.x;          // 0..255 -> (head = t/32, ch = t%32)
    int h = t >> 5;
    int s, d; edge_sd(ei, E, e, s, d);
    float v = asrc[(size_t)s * HEADS + h] + adst[(size_t)d * HEADS + h];
    v = v > 0.0f ? v : NEG_SLOPE * v;
    float alpha = __expf(v - emax[(size_t)d * HEADS + h]) / esum[(size_t)d * HEADS + h];
    atomicAdd(&out1[(size_t)d * HID + t], alpha * h1[(size_t)s * HID + t]);
}

// bias + ELU (writes result over h1, which is dead after aggregation)
__global__ void gat_elu(const float* __restrict__ out1, const float* __restrict__ b1,
                        float* __restrict__ hE, long long total) {
    long long gid = (long long)blockIdx.x * blockDim.x + threadIdx.x;
    if (gid >= total) return;
    float v = out1[gid] + b1[gid & (HID - 1)];
    hE[gid] = v > 0.0f ? v : __expf(v) - 1.0f;
}

// ---------------------------------- layer 2: GEMV 256->2 + attention logits
// one wave per node, lanes stride the 256 channels, shuffle reduction

__global__ void __launch_bounds__(256)
gat_lin2(const float* __restrict__ hE, const float* __restrict__ W2,
         const float* __restrict__ att_src2, const float* __restrict__ att_dst2,
         float* __restrict__ h2, float* __restrict__ asrc2, float* __restrict__ adst2,
         int n) {
    int wave = threadIdx.x >> 5, lane = threadIdx.x & 31;
    long long node = (long long)blockIdx.x * 8 + wave;
    if (node >= n) return;
    float s0 = 0.0f, s1 = 0.0f;
#pragma unroll
    for (int j = 0; j < HID / 32; ++j) {
        int k = lane + 32 * j;
        float hv = hE[(size_t)node * HID + k];
        s0 += hv * W2[(size_t)k * OUT_C + 0];
        s1 += hv * W2[(size_t)k * OUT_C + 1];
    }
#pragma unroll
    for (int off = 16; off > 0; off >>= 1) {
        s0 += __shfl_down(s0, off, 32);
        s1 += __shfl_down(s1, off, 32);
    }
    if (lane == 0) {
        h2[(size_t)node * OUT_C + 0] = s0;
        h2[(size_t)node * OUT_C + 1] = s1;
        asrc2[node] = s0 * att_src2[0] + s1 * att_src2[1];
        adst2[node] = s0 * att_dst2[0] + s1 * att_dst2[1];
    }
}

// --------------------------------------------- layer-2 softmax + aggregation

__global__ void gat_e2_max(const long long* __restrict__ ei, long long E, long long Et,
                           const float* __restrict__ asrc, const float* __restrict__ adst,
                           float* __restrict__ emax) {
    long long e = (long long)blockIdx.x * blockDim.x + threadIdx.x;
    if (e >= Et) return;
    int s, d; edge_sd(ei, E, e, s, d);
    float v = asrc[s] + adst[d];
    v = v > 0.0f ? v : NEG_SLOPE * v;
    atomicMaxF(&emax[d], v);
}

__global__ void gat_e2_sum(const long long* __restrict__ ei, long long E, long long Et,
                           const float* __restrict__ asrc, const float* __restrict__ adst,
                           const float* __restrict__ emax, float* __restrict__ esum) {
    long long e = (long long)blockIdx.x * blockDim.x + threadIdx.x;
    if (e >= Et) return;
    int s, d; edge_sd(ei, E, e, s, d);
    float v = asrc[s] + adst[d];
    v = v > 0.0f ? v : NEG_SLOPE * v;
    atomicAdd(&esum[d], __expf(v - emax[d]));
}

__global__ void gat_e2_agg(const long long* __restrict__ ei, long long E, long long Et,
                           const float* __restrict__ asrc, const float* __restrict__ adst,
                           const float* __restrict__ emax, const float* __restrict__ esum,
                           const float* __restrict__ h2, float* __restrict__ out2) {
    long long e = (long long)blockIdx.x * blockDim.x + threadIdx.x;
    if (e >= Et) return;
    int s, d; edge_sd(ei, E, e, s, d);
    float v = asrc[s] + adst[d];
    v = v > 0.0f ? v : NEG_SLOPE * v;
    float alpha = __expf(v - emax[d]) / esum[d];
    atomicAdd(&out2[(size_t)d * OUT_C + 0], alpha * h2[(size_t)s * OUT_C + 0]);
    atomicAdd(&out2[(size_t)d * OUT_C + 1], alpha * h2[(size_t)s * OUT_C + 1]);
}

__global__ void gat_final(const float* __restrict__ out2, const float* __restrict__ b2,
                          float* __restrict__ out, long long total) {
    long long gid = (long long)blockIdx.x * blockDim.x + threadIdx.x;
    if (gid >= total) return;
    out[gid] = out2[gid] + b2[gid & 1];
}

// ---------------------------------------------------------------- host side

static inline size_t au(size_t x) { return (x + 255) & ~(size_t)255; }
static inline unsigned gdiv(long long n, int b) { return (unsigned)((n + b - 1) / b); }

extern "C" void kernel_launch(void* const* d_in, const int* in_sizes, int n_in,
                              void* d_out, int out_size, void* d_ws, size_t ws_size,
                              hipStream_t stream) {
    const float*     x        = (const float*)d_in[0];
    const long long* ei       = (const long long*)d_in[1];   // int64 [2, E]
    const float*     W1       = (const float*)d_in[2];
    const float*     att_src1 = (const float*)d_in[3];
    const float*     att_dst1 = (const float*)d_in[4];
    const float*     b1       = (const float*)d_in[5];
    const float*     W2       = (const float*)d_in[6];
    const float*     att_src2 = (const float*)d_in[7];
    const float*     att_dst2 = (const float*)d_in[8];
    const float*     b2       = (const float*)d_in[9];
    float*           out      = (float*)d_out;

    const int       N    = in_sizes[0] / IN_C;
    const long long E    = in_sizes[1] / 2;
    const long long Et   = E + N;
    const int       Mp16 = (N + 15) / 16 * 16;

    // workspace carve-up
    char* w = (char*)d_ws;
    size_t off = 0;
    _Float16* xh   = (_Float16*)(w + off); off = au(off + (size_t)Mp16 * KPAD * 2);
    _Float16* wt   = (_Float16*)(w + off); off = au(off + (size_t)HID * KPAD * 2);
    float* h1      = (float*)(w + off);    off = au(off + (size_t)N * HID * 4);
    float* asrc1   = (float*)(w + off);    off = au(off + (size_t)N * HEADS * 4);
    float* adst1   = (float*)(w + off);    off = au(off + (size_t)N * HEADS * 4);
    float* emax1   = (float*)(w + off);    off = au(off + (size_t)N * HEADS * 4);
    float* esum1   = (float*)(w + off);    off = au(off + (size_t)N * HEADS * 4);
    float* out1    = (float*)(w + off);    off = au(off + (size_t)N * HID * 4);
    float* h2      = (float*)(w + off);    off = au(off + (size_t)N * OUT_C * 4);
    float* asrc2   = (float*)(w + off);    off = au(off + (size_t)N * 4);
    float* adst2   = (float*)(w + off);    off = au(off + (size_t)N * 4);
    float* emax2   = (float*)(w + off);    off = au(off + (size_t)N * 4);
    float* esum2   = (float*)(w + off);    off = au(off + (size_t)N * 4);
    float* out2    = (float*)(w + off);    off = au(off + (size_t)N * OUT_C * 4);

    const float NEG_INF = -INFINITY;

    // init accumulators
    gat_fill<<<gdiv((long long)N * HID, 256), 256, 0, stream>>>(out1, 0.0f, (long long)N * HID);
    gat_fill<<<gdiv((long long)N * HEADS, 256), 256, 0, stream>>>(esum1, 0.0f, (long long)N * HEADS);
    gat_fill<<<gdiv((long long)N * HEADS, 256), 256, 0, stream>>>(emax1, NEG_INF, (long long)N * HEADS);
    gat_fill<<<gdiv((long long)N * OUT_C, 256), 256, 0, stream>>>(out2, 0.0f, (long long)N * OUT_C);
    gat_fill<<<gdiv(N, 256), 256, 0, stream>>>(esum2, 0.0f, N);
    gat_fill<<<gdiv(N, 256), 256, 0, stream>>>(emax2, NEG_INF, N);

    // stage f16 operands, run WMMA GEMM
    gat_cvt_x<<<gdiv((long long)Mp16 * KPAD, 256), 256, 0, stream>>>(x, xh, N, (long long)Mp16 * KPAD);
    gat_cvt_w<<<gdiv((long long)HID * KPAD, 256), 256, 0, stream>>>(W1, wt);
    {
        dim3 grid(Mp16 / 16, HID / 16);
        gat_gemm1<<<grid, 32, 0, stream>>>(xh, wt, h1, N);
    }

    // layer-1 attention
    gat_att1<<<gdiv((long long)N * HEADS, 256), 256, 0, stream>>>(h1, att_src1, att_dst1, asrc1, adst1, N);
    gat_e1_max<<<gdiv(Et * HEADS, 256), 256, 0, stream>>>(ei, E, Et, asrc1, adst1, emax1);
    gat_e1_sum<<<gdiv(Et * HEADS, 256), 256, 0, stream>>>(ei, E, Et, asrc1, adst1, emax1, esum1);
    gat_e1_agg<<<(unsigned)Et, 256, 0, stream>>>(ei, E, Et, asrc1, adst1, emax1, esum1, h1, out1);

    // bias + ELU (result overwrites h1)
    gat_elu<<<gdiv((long long)N * HID, 256), 256, 0, stream>>>(out1, b1, h1, (long long)N * HID);

    // layer-2 linear + attention
    gat_lin2<<<gdiv(N, 8), 256, 0, stream>>>(h1, W2, att_src2, att_dst2, h2, asrc2, adst2, N);
    gat_e2_max<<<gdiv(Et, 256), 256, 0, stream>>>(ei, E, Et, asrc2, adst2, emax2);
    gat_e2_sum<<<gdiv(Et, 256), 256, 0, stream>>>(ei, E, Et, asrc2, adst2, emax2, esum2);
    gat_e2_agg<<<gdiv(Et, 256), 256, 0, stream>>>(ei, E, Et, asrc2, adst2, emax2, esum2, h2, out2);

    gat_final<<<gdiv((long long)N * OUT_C, 256), 256, 0, stream>>>(out2, b2, out, (long long)N * OUT_C);
}
